// TransformerDecoder_7267084665457
// MI455X (gfx1250) — matrix operand order repaired and checked by reading, not compiled, not run on hardware
//
#include <hip/hip_runtime.h>

// ---------------------------------------------------------------------------
// Types for CDNA5 WMMA (gfx1250, wave32)
// ---------------------------------------------------------------------------
typedef _Float16 v8h  __attribute__((ext_vector_type(8)));
typedef _Float16 v16h __attribute__((ext_vector_type(16)));
typedef float    v8f  __attribute__((ext_vector_type(8)));

#define DEVINL __device__ __forceinline__

// Problem constants (from reference)
static constexpr int NB_  = 8;     // batch
static constexpr int TX_  = 512;
static constexpr int TY_  = 1024;
static constexpr int DD_  = 512;   // model dim (= K = N of every GEMM)
static constexpr int HH_  = 8;     // heads
static constexpr float NEGF = -1.0e30f;

// ---------------------------------------------------------------------------
// WMMA fragment loaders (layouts per CDNA5 ISA §7.12.2)
//
// A-matrix 16x32 f16: lane l -> row = l&15, half g = l>>4:
//   v16h elems 0..7  = K[g*8 .. g*8+7],  elems 8..15 = K[16+g*8 .. 16+g*8+7]
// B-matrix 32x16 f16 (stored transposed in LDS as Bt[n][k]): lane l ->
//   col n = l&15, elems 0..7 = K[g*8 .. g*8+7], elems 8..15 = K[16+g*8 .. 16+g*8+7]
// Both halves are contiguous runs in LDS -> ds_load_b128 pairs.
// ---------------------------------------------------------------------------
DEVINL v16h ld_afrag(const _Float16* p, int stride, int lane) {
  const int r = lane & 15, g = lane >> 4;
  const _Float16* q = p + r * stride + g * 8;
  v8h lo = *(const v8h*)(q);
  v8h hi = *(const v8h*)(q + 16);
  return __builtin_shufflevector(lo, hi, 0,1,2,3,4,5,6,7,8,9,10,11,12,13,14,15);
}
DEVINL v16h ld_bfrag(const _Float16* p, int stride, int lane) {
  const int n = lane & 15, g = lane >> 4;
  const _Float16* q = p + n * stride + g * 8;
  v8h lo = *(const v8h*)(q);
  v8h hi = *(const v8h*)(q + 16);
  return __builtin_shufflevector(lo, hi, 0,1,2,3,4,5,6,7,8,9,10,11,12,13,14,15);
}
DEVINL v8f wmma_f16(v16h a, v16h b, v8f c) {
  return __builtin_amdgcn_wmma_f32_16x16x32_f16(false, a, false, b, (short)0, c,
                                                false, false);
}

// ---------------------------------------------------------------------------
// fp32 -> fp16 conversion (weights)
// ---------------------------------------------------------------------------
__global__ void cvt_f32_f16_kernel(const float* __restrict__ in,
                                   _Float16* __restrict__ out, int n) {
  int i = blockIdx.x * blockDim.x + threadIdx.x;
  const int stride = gridDim.x * blockDim.x;
  for (; i < n; i += stride) out[i] = (_Float16)in[i];
}

// ---------------------------------------------------------------------------
// LayerNorm + ReLU -> fp16.  One block (128 threads) per row of 512.
// ---------------------------------------------------------------------------
__global__ __launch_bounds__(128)
void ln_relu_kernel(const float* __restrict__ X, const float* __restrict__ g,
                    const float* __restrict__ bta, _Float16* __restrict__ Y) {
  __shared__ float sm[4];
  const int row = blockIdx.x;
  const int t = threadIdx.x, lane = t & 31, wv = t >> 5;
  const float* x = X + (size_t)row * DD_;
  float v[4];
#pragma unroll
  for (int j = 0; j < 4; ++j) v[j] = x[t + 128 * j];
  float s = v[0] + v[1] + v[2] + v[3];
#pragma unroll
  for (int off = 16; off; off >>= 1) s += __shfl_xor(s, off, 32);
  if (lane == 0) sm[wv] = s;
  __syncthreads();
  const float mean = (sm[0] + sm[1] + sm[2] + sm[3]) * (1.0f / 512.0f);
  float s2 = 0.0f;
#pragma unroll
  for (int j = 0; j < 4; ++j) { float d = v[j] - mean; s2 += d * d; }
#pragma unroll
  for (int off = 16; off; off >>= 1) s2 += __shfl_xor(s2, off, 32);
  __syncthreads();
  if (lane == 0) sm[wv] = s2;
  __syncthreads();
  const float var = (sm[0] + sm[1] + sm[2] + sm[3]) * (1.0f / 512.0f);
  const float rstd = rsqrtf(var + 1e-5f);
#pragma unroll
  for (int j = 0; j < 4; ++j) {
    const int c = t + 128 * j;
    const float o = (v[j] - mean) * rstd * g[c] + bta[c];
    Y[(size_t)row * DD_ + c] = (_Float16)fmaxf(o, 0.0f);
  }
}

// ---------------------------------------------------------------------------
// GEMM:  C[M,512] = A[M,512] @ W[512,512] + bias, epilogue variants.
// Block = 128 threads (4 waves); 128x64 C tile; wave w -> rows 32w..32w+31,
// all 64 cols (2x4 accumulators).  K staged 64-wide per barrier pair; per
// wave that is 16 WMMAs fed by 24 ds_load_b128 (B-frags reused across two
// A row-groups) -- keeps the matrix pipe ahead of LDS bandwidth.
//   OMODE 0: fp32 out            OMODE 1: fp16 out
//   OMODE 2: fp32 out + residual OMODE 3: fp32 out + row-masked + residual
// ---------------------------------------------------------------------------
template <bool A32, int OMODE>
__global__ __launch_bounds__(128)
void gemm512_kernel(const void* __restrict__ Ap, const _Float16* __restrict__ Wp,
                    const float* __restrict__ bias, float* __restrict__ C32,
                    _Float16* __restrict__ C16, const float* __restrict__ R,
                    const unsigned char* __restrict__ rowmask) {
  constexpr int SP = 72;  // LDS row stride (fp16): 144B, keeps b128 alignment
  __shared__ _Float16 As[128 * SP];  // 128 rows x 64 K
  __shared__ _Float16 Bt[64 * SP];   // transposed: Bt[n][k], 64 n x 64 k
  const int t = threadIdx.x;
  const int w = t >> 5, lane = t & 31, half = lane >> 4, cn = lane & 15;
  const int m0 = blockIdx.y * 128, n0 = blockIdx.x * 64;

  v8f acc[2][4] = {};

  for (int kb = 0; kb < DD_; kb += 64) {
    __syncthreads();
    // ---- stage A tile (128 rows x 64 K): one row per thread ----
    {
      const int r = t;
      if constexpr (A32) {
        const float* ar = (const float*)Ap + (size_t)(m0 + r) * DD_ + kb;
#pragma unroll
        for (int q = 0; q < 16; ++q) {
          float4 f = ((const float4*)ar)[q];
          _Float16* d = &As[r * SP + q * 4];
          d[0] = (_Float16)f.x; d[1] = (_Float16)f.y;
          d[2] = (_Float16)f.z; d[3] = (_Float16)f.w;
        }
      } else {
        const _Float16* ar = (const _Float16*)Ap + (size_t)(m0 + r) * DD_ + kb;
        int4* dst = (int4*)&As[r * SP];
#pragma unroll
        for (int q = 0; q < 8; ++q) dst[q] = ((const int4*)ar)[q];
      }
    }
    // ---- stage W tile transposed: Bt[n][k], 64 k x 64 n ----
    {
      const int k = t >> 1, c0 = (t & 1) * 32;
      const _Float16* wr = Wp + (size_t)(kb + k) * DD_ + n0 + c0;
#pragma unroll
      for (int q = 0; q < 4; ++q) {
        v8h wv = ((const v8h*)wr)[q];
#pragma unroll
        for (int j = 0; j < 8; ++j) Bt[(c0 + q * 8 + j) * SP + k] = wv[j];
      }
    }
    __syncthreads();
#pragma unroll
    for (int ks = 0; ks < 64; ks += 32) {
      const v16h a0 = ld_afrag(&As[(32 * w) * SP + ks], SP, lane);
      const v16h a1 = ld_afrag(&As[(32 * w + 16) * SP + ks], SP, lane);
#pragma unroll
      for (int tt = 0; tt < 4; ++tt) {
        const v16h b = ld_bfrag(&Bt[(16 * tt) * SP + ks], SP, lane);
        acc[0][tt] = wmma_f16(a0, b, acc[0][tt]);
        acc[1][tt] = wmma_f16(a1, b, acc[1][tt]);
      }
    }
  }

  // ---- epilogue: C layout -> lane = col, VGPR i + 8*half = row ----
#pragma unroll
  for (int ah = 0; ah < 2; ++ah)
#pragma unroll
    for (int tt = 0; tt < 4; ++tt) {
      const int col = n0 + 16 * tt + cn;
      const float bv = bias[col];
#pragma unroll
      for (int i = 0; i < 8; ++i) {
        const int row = m0 + 32 * w + 16 * ah + i + 8 * half;
        const size_t idx = (size_t)row * DD_ + col;
        const float v = acc[ah][tt][i] + bv;
        if constexpr (OMODE == 0)      C32[idx] = v;
        else if constexpr (OMODE == 1) C16[idx] = (_Float16)v;
        else if constexpr (OMODE == 2) C32[idx] = R[idx] + v;
        else                           C32[idx] = R[idx] + (rowmask[row] ? v : 0.0f);
      }
    }
}

// ---------------------------------------------------------------------------
// Flash attention (key-masked softmax).  Grid = (TQ/64, H, B), 128 threads.
// Each wave owns 16 query rows; S = Q K^T and O += P V both via WMMA.
// Query-side masking is omitted (masked-query outputs are discarded by the
// masked-residual epilogue, matching the reference semantics).
// ---------------------------------------------------------------------------
__global__ __launch_bounds__(128)
void attn_kernel(const _Float16* __restrict__ Q, const _Float16* __restrict__ Kt,
                 const _Float16* __restrict__ V, const unsigned char* __restrict__ mk,
                 _Float16* __restrict__ O, int TQ, int TK) {
  constexpr int SP = 72;  // 144B stride, 16B-aligned
  __shared__ _Float16 Qs[64 * SP];
  __shared__ _Float16 Ks[64 * SP];
  __shared__ _Float16 Vt[64 * SP];     // transposed: Vt[d][key]
  __shared__ _Float16 Ps[4][16 * SP];  // per-wave P in A-layout
  __shared__ float    mskf[64];

  const int b = blockIdx.z, h = blockIdx.y;
  const int qbase = blockIdx.x * 64;
  const int t = threadIdx.x, w = t >> 5, lane = t & 31;
  const int half = lane >> 4, cn = lane & 15;

  // stage Q tile (64 rows x 64 d)
  {
    const int r = t >> 1, c0 = (t & 1) * 32;
    const _Float16* src = Q + ((size_t)(b * TQ + qbase + r)) * DD_ + h * 64 + c0;
    int4* dst = (int4*)&Qs[r * SP + c0];
    const int4* s4 = (const int4*)src;
    dst[0] = s4[0]; dst[1] = s4[1]; dst[2] = s4[2]; dst[3] = s4[3];
  }

  v8f o[4] = {};
  float mst[8], lst[8];
#pragma unroll
  for (int i = 0; i < 8; ++i) { mst[i] = -3.0e38f; lst[i] = 0.0f; }

  for (int kb = 0; kb < TK; kb += 64) {
    __syncthreads();  // previous tile's WMMA reads of Ks/Vt done
    // ---- stage K tile row-major, V tile transposed, mask ----
    {
      const int r = t >> 1, c0 = (t & 1) * 32;
      const _Float16* ks = Kt + ((size_t)(b * TK + kb + r)) * DD_ + h * 64 + c0;
      int4* dst = (int4*)&Ks[r * SP + c0];
      const int4* s4 = (const int4*)ks;
      dst[0] = s4[0]; dst[1] = s4[1]; dst[2] = s4[2]; dst[3] = s4[3];
      const _Float16* vs = V + ((size_t)(b * TK + kb + r)) * DD_ + h * 64 + c0;
#pragma unroll
      for (int q = 0; q < 4; ++q) {
        v8h vv = ((const v8h*)vs)[q];
#pragma unroll
        for (int j = 0; j < 8; ++j) Vt[(c0 + q * 8 + j) * SP + r] = vv[j];
      }
    }
    if (t < 64) mskf[t] = mk[b * TK + kb + t] ? 0.0f : NEGF;
    __syncthreads();

    // ---- S = Q K^T (per wave: 16 rows x 64 keys) ----
    v8f s[4] = {};
#pragma unroll
    for (int ks2 = 0; ks2 < 64; ks2 += 32) {
      const v16h a = ld_afrag(&Qs[(16 * w) * SP + ks2], SP, lane);
#pragma unroll
      for (int tt = 0; tt < 4; ++tt) {
        const v16h bb = ld_bfrag(&Ks[(16 * tt) * SP + ks2], SP, lane);
        s[tt] = wmma_f16(a, bb, s[tt]);
      }
    }

    // ---- online softmax (fp32), stats per row within 16-lane half ----
    float sv[4][8], rmax[8];
#pragma unroll
    for (int i = 0; i < 8; ++i) rmax[i] = -3.0e38f;
#pragma unroll
    for (int tt = 0; tt < 4; ++tt) {
      const float ma = mskf[16 * tt + cn];
#pragma unroll
      for (int i = 0; i < 8; ++i) {
        const float x = s[tt][i] * 0.125f + ma;  // DENOM = 1/sqrt(64)
        sv[tt][i] = x;
        rmax[i] = fmaxf(rmax[i], x);
      }
    }
#pragma unroll
    for (int off = 8; off; off >>= 1)
#pragma unroll
      for (int i = 0; i < 8; ++i) rmax[i] = fmaxf(rmax[i], __shfl_xor(rmax[i], off, 32));

    float corr[8], rsum[8];
#pragma unroll
    for (int i = 0; i < 8; ++i) {
      const float mn = fmaxf(mst[i], rmax[i]);
      corr[i] = __expf(mst[i] - mn);
      mst[i] = mn;
      rsum[i] = 0.0f;
    }
#pragma unroll
    for (int tt = 0; tt < 4; ++tt)
#pragma unroll
      for (int i = 0; i < 8; ++i) {
        const float p = __expf(sv[tt][i] - mst[i]);
        rsum[i] += p;
        Ps[w][(i + 8 * half) * SP + 16 * tt + cn] = (_Float16)p;
      }
#pragma unroll
    for (int off = 8; off; off >>= 1)
#pragma unroll
      for (int i = 0; i < 8; ++i) rsum[i] += __shfl_xor(rsum[i], off, 32);
#pragma unroll
    for (int i = 0; i < 8; ++i) lst[i] = lst[i] * corr[i] + rsum[i];
#pragma unroll
    for (int tt = 0; tt < 4; ++tt)
#pragma unroll
      for (int i = 0; i < 8; ++i) o[tt][i] *= corr[i];

    __syncthreads();  // Ps visible to all lanes of the wave

    // ---- O += P @ V ----
#pragma unroll
    for (int ks2 = 0; ks2 < 64; ks2 += 32) {
      const v16h a = ld_afrag(&Ps[w][ks2], SP, lane);
#pragma unroll
      for (int tt = 0; tt < 4; ++tt) {
        const v16h bb = ld_bfrag(&Vt[(16 * tt) * SP + ks2], SP, lane);
        o[tt] = wmma_f16(a, bb, o[tt]);
      }
    }
  }

  // ---- finalize: O / l, write fp16 ----
#pragma unroll
  for (int tt = 0; tt < 4; ++tt)
#pragma unroll
    for (int i = 0; i < 8; ++i) {
      const int row = b * TQ + qbase + 16 * w + i + 8 * half;
      const int col = h * 64 + 16 * tt + cn;
      O[(size_t)row * DD_ + col] = (_Float16)(o[tt][i] / lst[i]);
    }
}

// ---------------------------------------------------------------------------
// Final output: out = mask_x ? x : 0
// ---------------------------------------------------------------------------
__global__ void mask_out_kernel(const float* __restrict__ X,
                                const unsigned char* __restrict__ m,
                                float* __restrict__ out, int total) {
  const int i = blockIdx.x * blockDim.x + threadIdx.x;
  if (i < total) out[i] = m[i >> 9] ? X[i] : 0.0f;
}

// ---------------------------------------------------------------------------
// Host-side parameter plumbing
// ---------------------------------------------------------------------------
struct ResP { const float *g1, *b1, *W1, *c1, *g2, *b2, *W2, *c2; };
struct AttP { const float *wq, *bq, *wk, *bk, *wv, *bv, *wo, *bo; };
struct AllP {
  const float *x, *y;
  const unsigned char *mx, *my;
  ResP rb[2], ra[2];
  AttP sa[3], ca[3];
  const float *ln1g[3], *ln1b[3], *ln2g[3], *ln2b[3];
};

static ResP res_ins(void* const* d, int b0) {  // g1,b1,W1,c1,g2,b2,W2,c2
  return ResP{(const float*)d[b0+0],(const float*)d[b0+1],(const float*)d[b0+2],
              (const float*)d[b0+3],(const float*)d[b0+4],(const float*)d[b0+5],
              (const float*)d[b0+6],(const float*)d[b0+7]};
}
static ResP res_sorted(void* const* d, int b0) {  // W1,W2,b1,b2,c1,c2,g1,g2
  return ResP{(const float*)d[b0+6],(const float*)d[b0+2],(const float*)d[b0+0],
              (const float*)d[b0+4],(const float*)d[b0+7],(const float*)d[b0+3],
              (const float*)d[b0+1],(const float*)d[b0+5]};
}
static AttP att_ins(void* const* d, int b0) {  // wq,bq,wk,bk,wv,bv,wo,bo
  return AttP{(const float*)d[b0+0],(const float*)d[b0+1],(const float*)d[b0+2],
              (const float*)d[b0+3],(const float*)d[b0+4],(const float*)d[b0+5],
              (const float*)d[b0+6],(const float*)d[b0+7]};
}
static AttP att_sorted(void* const* d, int b0) {  // bk,bo,bq,bv,wk,wo,wq,wv
  return AttP{(const float*)d[b0+6],(const float*)d[b0+2],(const float*)d[b0+4],
              (const float*)d[b0+0],(const float*)d[b0+7],(const float*)d[b0+3],
              (const float*)d[b0+5],(const float*)d[b0+1]};
}

static void fill_params(AllP& P, void* const* d, const int* sz) {
  if (sz[0] == NB_ * TX_) {
    // fully sorted pytree: mask_x, mask_y, params(sorted), x, y
    P.mx = (const unsigned char*)d[0];
    P.my = (const unsigned char*)d[1];
    const int pb = 2;
    for (int l = 0; l < 3; ++l) P.ca[l] = att_sorted(d, pb + 8 * l);
    for (int l = 0; l < 3; ++l) { P.ln1b[l] = (const float*)d[pb+24+l];
                                  P.ln1g[l] = (const float*)d[pb+27+l];
                                  P.ln2b[l] = (const float*)d[pb+30+l];
                                  P.ln2g[l] = (const float*)d[pb+33+l]; }
    for (int i = 0; i < 2; ++i) P.ra[i] = res_sorted(d, pb + 36 + 8 * i);
    for (int i = 0; i < 2; ++i) P.rb[i] = res_sorted(d, pb + 52 + 8 * i);
    for (int l = 0; l < 3; ++l) P.sa[l] = att_sorted(d, pb + 68 + 8 * l);
    P.x = (const float*)d[94]; P.y = (const float*)d[95];
  } else if (sz[6] == DD_ * DD_) {
    // insertion order everywhere
    P.x = (const float*)d[0]; P.y = (const float*)d[1];
    P.mx = (const unsigned char*)d[2]; P.my = (const unsigned char*)d[3];
    for (int i = 0; i < 2; ++i) P.rb[i] = res_ins(d, 4 + 8 * i);
    for (int i = 0; i < 2; ++i) P.ra[i] = res_ins(d, 20 + 8 * i);
    for (int l = 0; l < 3; ++l) P.sa[l] = att_ins(d, 36 + 8 * l);
    for (int l = 0; l < 3; ++l) P.ca[l] = att_ins(d, 60 + 8 * l);
    for (int l = 0; l < 3; ++l) { P.ln1g[l] = (const float*)d[84+l];
                                  P.ln1b[l] = (const float*)d[87+l];
                                  P.ln2g[l] = (const float*)d[90+l];
                                  P.ln2b[l] = (const float*)d[93+l]; }
  } else {
    // top-level insertion order, params dict sorted
    P.x = (const float*)d[0]; P.y = (const float*)d[1];
    P.mx = (const unsigned char*)d[2]; P.my = (const unsigned char*)d[3];
    for (int l = 0; l < 3; ++l) P.ca[l] = att_sorted(d, 4 + 8 * l);
    for (int l = 0; l < 3; ++l) { P.ln1b[l] = (const float*)d[28+l];
                                  P.ln1g[l] = (const float*)d[31+l];
                                  P.ln2b[l] = (const float*)d[34+l];
                                  P.ln2g[l] = (const float*)d[37+l]; }
    for (int i = 0; i < 2; ++i) P.ra[i] = res_sorted(d, 40 + 8 * i);
    for (int i = 0; i < 2; ++i) P.rb[i] = res_sorted(d, 56 + 8 * i);
    for (int l = 0; l < 3; ++l) P.sa[l] = att_sorted(d, 72 + 8 * l);
  }
}

// ---------------------------------------------------------------------------
// Launch helpers
// ---------------------------------------------------------------------------
static void cvt_w(const float* src, _Float16* dst, hipStream_t s) {
  cvt_f32_f16_kernel<<<512, 256, 0, s>>>(src, dst, DD_ * DD_);
}

template <bool A32, int OMODE>
static void gemm(const void* A, const _Float16* W, const float* bias, void* C,
                 const float* R, const unsigned char* mask, int M, hipStream_t s) {
  gemm512_kernel<A32, OMODE><<<dim3(DD_ / 64, M / 128), 128, 0, s>>>(
      A, W, bias, (float*)C, (_Float16*)C, R, mask);
}

static void resblock(float* Xb, int rows, const ResP& rp, _Float16* A16,
                     float* H32, _Float16* W16, hipStream_t s) {
  ln_relu_kernel<<<rows, 128, 0, s>>>(Xb, rp.g1, rp.b1, A16);
  cvt_w(rp.W1, W16, s);
  gemm<false, 0>(A16, W16, rp.c1, H32, nullptr, nullptr, rows, s);
  ln_relu_kernel<<<rows, 128, 0, s>>>(H32, rp.g2, rp.b2, A16);
  cvt_w(rp.W2, W16, s);
  gemm<false, 2>(A16, W16, rp.c2, Xb, Xb, nullptr, rows, s);
}

// ---------------------------------------------------------------------------
// kernel_launch
// ---------------------------------------------------------------------------
extern "C" void kernel_launch(void* const* d_in, const int* in_sizes, int n_in,
                              void* d_out, int out_size, void* d_ws, size_t ws_size,
                              hipStream_t stream) {
  if (n_in < 96) return;
  AllP P;
  fill_params(P, d_in, in_sizes);

  const int RY = NB_ * TY_;  // 8192 rows
  const int RX = NB_ * TX_;  // 4096 rows

  // workspace layout
  char* p = (char*)d_ws;
  float*    yb  = (float*)p;    p += (size_t)RY * DD_ * 4;   // 16 MB
  float*    xb  = (float*)p;    p += (size_t)RX * DD_ * 4;   //  8 MB
  float*    H32 = (float*)p;    p += (size_t)RY * DD_ * 4;   // 16 MB
  _Float16* A16 = (_Float16*)p; p += (size_t)RY * DD_ * 2;   //  8 MB
  _Float16* Q16 = (_Float16*)p; p += (size_t)RY * DD_ * 2;   //  8 MB
  _Float16* K16 = (_Float16*)p; p += (size_t)RY * DD_ * 2;   //  8 MB
  _Float16* V16 = (_Float16*)p; p += (size_t)RY * DD_ * 2;   //  8 MB
  _Float16* W16 = (_Float16*)p; p += (size_t)DD_ * DD_ * 2;  // 0.5 MB
  (void)ws_size;

  hipMemcpyAsync(yb, P.y, (size_t)RY * DD_ * 4, hipMemcpyDeviceToDevice, stream);
  hipMemcpyAsync(xb, P.x, (size_t)RX * DD_ * 4, hipMemcpyDeviceToDevice, stream);

  // ---- res_before blocks on y ----
  for (int i = 0; i < 2; ++i) resblock(yb, RY, P.rb[i], A16, H32, W16, stream);

  // ---- self-attention layers on y ----
  for (int l = 0; l < 3; ++l) {
    ln_relu_kernel<<<RY, 128, 0, stream>>>(yb, P.ln1g[l], P.ln1b[l], A16);
    cvt_w(P.sa[l].wq, W16, stream);
    gemm<false, 1>(A16, W16, P.sa[l].bq, Q16, nullptr, nullptr, RY, stream);
    cvt_w(P.sa[l].wk, W16, stream);
    gemm<false, 1>(A16, W16, P.sa[l].bk, K16, nullptr, nullptr, RY, stream);
    cvt_w(P.sa[l].wv, W16, stream);
    gemm<false, 1>(A16, W16, P.sa[l].bv, V16, nullptr, nullptr, RY, stream);
    attn_kernel<<<dim3(TY_ / 64, HH_, NB_), 128, 0, stream>>>(
        Q16, K16, V16, P.my, A16, TY_, TY_);
    cvt_w(P.sa[l].wo, W16, stream);
    gemm<false, 3>(A16, W16, P.sa[l].bo, yb, yb, P.my, RY, stream);
  }

  // ---- cross-attention layers: q from x, k/v from raw y ----
  for (int l = 0; l < 3; ++l) {
    ln_relu_kernel<<<RX, 128, 0, stream>>>(xb, P.ln2g[l], P.ln2b[l], A16);
    cvt_w(P.ca[l].wq, W16, stream);
    gemm<false, 1>(A16, W16, P.ca[l].bq, Q16, nullptr, nullptr, RX, stream);
    cvt_w(P.ca[l].wk, W16, stream);
    gemm<true, 1>(yb, W16, P.ca[l].bk, K16, nullptr, nullptr, RY, stream);
    cvt_w(P.ca[l].wv, W16, stream);
    gemm<true, 1>(yb, W16, P.ca[l].bv, V16, nullptr, nullptr, RY, stream);
    attn_kernel<<<dim3(TX_ / 64, HH_, NB_), 128, 0, stream>>>(
        Q16, K16, V16, P.my, A16, TX_, TY_);
    cvt_w(P.ca[l].wo, W16, stream);
    gemm<false, 3>(A16, W16, P.ca[l].bo, xb, xb, P.mx, RX, stream);
  }

  // ---- res_after blocks on x ----
  for (int i = 0; i < 2; ++i) resblock(xb, RX, P.ra[i], A16, H32, W16, stream);

  // ---- final mask ----
  const int total = RX * DD_;
  mask_out_kernel<<<total / 256, 256, 0, stream>>>(xb, P.mx, (float*)d_out, total);
}
